// TrainableMLA_80453327389024
// MI455X (gfx1250) — matrix operand order repaired and checked by reading, not compiled, not run on hardware
//
#include <hip/hip_runtime.h>
#include <hip/hip_bf16.h>

// MLA forward for MI455X (gfx1250). All matmuls via v_wmma_f32_16x16x32_bf16
// (wave32, fp32 accumulate); softmax / RMSNorm / RoPE in fp32.
//
// LDS layouts make every WMMA fragment a contiguous 32-byte run per lane
// (-> two ds_load_b128) and all global traffic float4 (b128):
//  - A tiles: contiguous K [m][k] (no permutation), fragment =
//    As[m][16*hi .. 16*hi+15].
//  - B / V tiles: transposed [n][k]; fragment = BsT[n][16*hi .. 16*hi+15].
//    Transposed staging packs (k, k+1) pairs with one v_cvt_pk_bf16_f32 and
//    stores them as a single ds_store_b32 (no scalar b16 scatter).
//  - C/D 16x16 f32: lane -> col N = lane&15, VGPR v -> row M = v + 8*(lane>>4).
//
// Workspace (floats), total 4096*10368 ~= 170 MB:
//   q_a[4096x512] kv[4096x576] q[4096x3072] kv_up[4096x4096] kpe[4096x64] attn[4096x2048]

typedef __attribute__((ext_vector_type(2)))  __bf16 bf16x2;
typedef __attribute__((ext_vector_type(4)))  __bf16 bf16x4;
typedef __attribute__((ext_vector_type(8)))  __bf16 bf16x8;
typedef __attribute__((ext_vector_type(16))) __bf16 bf16x16;
typedef __attribute__((ext_vector_type(8)))  float  f32x8;

#define WMMA_BF16(A_, B_, C_) \
  __builtin_amdgcn_wmma_f32_16x16x32_bf16(false, (A_), false, (B_), (short)0, (C_), false, false)

__device__ __forceinline__ bf16x2 pk2(float a, float b) {
  bf16x2 r; r[0] = (__bf16)a; r[1] = (__bf16)b;  // one v_cvt_pk_bf16_f32
  return r;
}
__device__ __forceinline__ bf16x4 cvt4(float4 f) {
  bf16x4 r; r[0] = (__bf16)f.x; r[1] = (__bf16)f.y; r[2] = (__bf16)f.z; r[3] = (__bf16)f.w;
  return r;
}
__device__ __forceinline__ bf16x8 cat8(bf16x4 a, bf16x4 b) {
  return __builtin_shufflevector(a, b, 0, 1, 2, 3, 4, 5, 6, 7);
}
__device__ __forceinline__ bf16x16 cat16(bf16x8 a, bf16x8 b) {
  return __builtin_shufflevector(a, b, 0, 1, 2, 3, 4, 5, 6, 7, 8, 9, 10, 11, 12, 13, 14, 15);
}
// Load 16 contiguous bf16 from LDS (16B-aligned) as a WMMA operand.
__device__ __forceinline__ bf16x16 ldsfrag(const __bf16* p) {
  bf16x8 lo  = *(const bf16x8*)p;
  bf16x8 hiv = *(const bf16x8*)(p + 8);
  return cat16(lo, hiv);
}

// =======================================================================
// GEMM: C[M,N] = A[M,K] @ B[K,N]; fp32 in/out, bf16 WMMA compute.
// 256 threads (8 waves), block tile 64x64, K-step 32; 2 WMMA/wave/step.
// Requires M%64==0, N%64==0, K%32==0, all strides %4==0 (true here).
// =======================================================================
__global__ __launch_bounds__(256)
void gemm_bf16_wmma(const float* __restrict__ A, const float* __restrict__ B,
                    float* __restrict__ C, int K, int lda, int ldb, int ldc)
{
  __shared__ __align__(16) __bf16 As[64][40];   // contiguous-K layout [m][k]
  __shared__ __align__(16) __bf16 BsT[64][40];  // transposed [n][k]

  const int tid  = threadIdx.x;
  const int wave = tid >> 5, lane = tid & 31;
  const int hi   = lane >> 4, ln  = lane & 15;
  const int wr   = wave >> 1, wc  = wave & 1;         // 4(M) x 2(N) waves
  const int m0   = blockIdx.y * 64, n0 = blockIdx.x * 64;

  // Staging: A = 64x32 (2 quads/thread); B = 32x64 (2 adjacent-k quads/thread).
  const int ai  = tid >> 3;                 // A row 0..31 (+32 for 2nd quad)
  const int aj  = (tid & 7) << 2;           // A k-quad start
  const int apj = aj;  // contiguous dest
  const int bk  = (tid >> 4) << 1;          // B k pair: 0,2,...,30
  const int bn  = (tid & 15) << 2;          // B n-quad start

  const float* ap0 = A + (size_t)(m0 + ai) * lda + aj;
  const float* ap1 = ap0 + (size_t)32 * lda;
  const float* bp0 = B + (size_t)bk * ldb + n0 + bn;
  const float* bp1 = bp0 + ldb;

  const f32x8 zero = {0.f, 0.f, 0.f, 0.f, 0.f, 0.f, 0.f, 0.f};
  f32x8 acc0 = zero, acc1 = zero;

  for (int kk = 0; kk < K; kk += 32) {
    float4 a0 = *(const float4*)ap0;
    float4 a1 = *(const float4*)ap1;
    float4 b0 = *(const float4*)bp0;
    float4 b1 = *(const float4*)bp1;
    ap0 += 32; ap1 += 32;
    bp0 += (size_t)32 * ldb; bp1 += (size_t)32 * ldb;
    if (kk + 32 < K) { __builtin_prefetch(ap0, 0, 1); __builtin_prefetch(bp0, 0, 1); }

    *(bf16x4*)&As[ai][apj]      = cvt4(a0);
    *(bf16x4*)&As[ai + 32][apj] = cvt4(a1);
    // Transposed B: pack (k, k+1) per n -> one b32 store each.
    *(bf16x2*)&BsT[bn + 0][bk] = pk2(b0.x, b1.x);
    *(bf16x2*)&BsT[bn + 1][bk] = pk2(b0.y, b1.y);
    *(bf16x2*)&BsT[bn + 2][bk] = pk2(b0.z, b1.z);
    *(bf16x2*)&BsT[bn + 3][bk] = pk2(b0.w, b1.w);
    __syncthreads();

    bf16x16 af  = ldsfrag(&As[wr * 16 + ln][16 * hi]);
    bf16x16 bf0 = ldsfrag(&BsT[wc * 32 + ln][16 * hi]);
    bf16x16 bf1 = ldsfrag(&BsT[wc * 32 + 16 + ln][16 * hi]);
    acc0 = WMMA_BF16(af, bf0, acc0);
    acc1 = WMMA_BF16(af, bf1, acc1);
    __syncthreads();
  }

  #pragma unroll
  for (int v = 0; v < 8; ++v) {
    float* crow = C + (size_t)(m0 + wr * 16 + v + 8 * hi) * ldc + n0 + wc * 32;
    crow[ln]      = acc0[v];
    crow[16 + ln] = acc1[v];
  }
}

// =======================================================================
// RMSNorm over 512 elements/row (in-place capable), fully float4.
// =======================================================================
__global__ __launch_bounds__(128)
void rmsnorm512_kernel(const float* __restrict__ in, const float* __restrict__ w,
                       float* __restrict__ out, int in_stride, int out_stride)
{
  const int row = blockIdx.x, tid = threadIdx.x;
  const float4 v = *(const float4*)(in + (size_t)row * in_stride + tid * 4);
  float ss = v.x * v.x + v.y * v.y + v.z * v.z + v.w * v.w;
  #pragma unroll
  for (int off = 16; off; off >>= 1) ss += __shfl_xor(ss, off);
  __shared__ float red[4];
  if ((tid & 31) == 0) red[tid >> 5] = ss;
  __syncthreads();
  float scale = rsqrtf((red[0] + red[1] + red[2] + red[3]) * (1.f / 512.f) + 1e-5f);
  const float4 wv = *(const float4*)(w + tid * 4);
  float4 o;
  o.x = v.x * scale * wv.x; o.y = v.y * scale * wv.y;
  o.z = v.z * scale * wv.z; o.w = v.w * scale * wv.w;
  *(float4*)(out + (size_t)row * out_stride + tid * 4) = o;
}

// =======================================================================
// RoPE: pairs (x[2i], x[2i+1]) -> (a*cos - b*sin, a*sin + b*cos)
// =======================================================================
__global__ __launch_bounds__(256)
void rope_q_kernel(float* __restrict__ q, const float* __restrict__ cosp,
                   const float* __restrict__ sinp)
{
  int idx = blockIdx.x * 256 + threadIdx.x;     // T * NH * 32 threads
  int i = idx & 31, h = (idx >> 5) & 15, tok = idx >> 9, s = tok & 2047;
  float c = cosp[s * 32 + i], sn = sinp[s * 32 + i];
  float* p = q + (size_t)tok * 3072 + h * 192 + 128 + 2 * i;
  float a = p[0], b = p[1];
  p[0] = a * c - b * sn;
  p[1] = a * sn + b * c;
}

__global__ __launch_bounds__(256)
void rope_k_kernel(const float* __restrict__ kv, float* __restrict__ kpe,
                   const float* __restrict__ cosp, const float* __restrict__ sinp)
{
  int idx = blockIdx.x * 256 + threadIdx.x;     // T * 32 threads
  int i = idx & 31, tok = idx >> 5, s = tok & 2047;
  float c = cosp[s * 32 + i], sn = sinp[s * 32 + i];
  const float* p = kv + (size_t)tok * 576 + 512 + 2 * i;
  float a = p[0], b = p[1];
  kpe[(size_t)tok * 64 + 2 * i]     = a * c - b * sn;
  kpe[(size_t)tok * 64 + 2 * i + 1] = a * sn + b * c;
}

// =======================================================================
// Causal flash attention: one (b,h) per blockIdx.y, 128 q-rows per block.
// 8 waves x 16 q-rows; 32-key tiles staged in LDS (K row-major, V transposed).
// Scores: 2 tiles x 6 WMMA (QK_DIM=192); PV: 8 WMMA (VDIM=128).
// =======================================================================
__global__ __launch_bounds__(256)
void mla_flash_attn(const float* __restrict__ q, const float* __restrict__ kvup,
                    const float* __restrict__ kpe, float* __restrict__ o)
{
  constexpr int   S     = 2048;
  constexpr float SCALE = 0.07216878364870322f;  // 192^-0.5

  __shared__ __align__(16) __bf16 Ks[32][208];    // [key][d], d-contiguous frags
  __shared__ __align__(16) __bf16 VsT[128][40];   // [vcol][key]
  __shared__ __align__(16) __bf16 Pb[8][16][40];  // per-wave 16x32 P, [row][key] contiguous

  const int tid  = threadIdx.x;
  const int wave = tid >> 5, lane = tid & 31;
  const int hi   = lane >> 4, ln  = lane & 15;
  const int bh   = blockIdx.y, b = bh >> 4, h = bh & 15;
  const int q0   = blockIdx.x * 128;
  const int qw   = q0 + wave * 16;          // wave's first q row (s-index)
  const int tok0 = b * S;

  // Q fragments (A layout): per chunk c, run d=[32c+16hi,+15].
  bf16x16 qf[6];
  {
    const float4* q4 = (const float4*)(q + (size_t)(tok0 + qw + ln) * 3072 + h * 192);
    #pragma unroll
    for (int c = 0; c < 6; ++c) {
      float4 r0 = q4[8 * c + 4 * hi];
      float4 r1 = q4[8 * c + 4 * hi + 1];
      float4 r2 = q4[8 * c + 4 * hi + 2];
      float4 r3 = q4[8 * c + 4 * hi + 3];
      qf[c] = cat16(cat8(cvt4(r0), cvt4(r1)), cat8(cvt4(r2), cvt4(r3)));
    }
  }

  const f32x8 zero = {0.f, 0.f, 0.f, 0.f, 0.f, 0.f, 0.f, 0.f};
  float m_i[8], l_i[8];
  f32x8 acc[8];
  #pragma unroll
  for (int v = 0; v < 8; ++v) { m_i[v] = -1e30f; l_i[v] = 0.f; }
  #pragma unroll
  for (int nt = 0; nt < 8; ++nt) acc[nt] = zero;

  // K staging: 8 threads/key, 6 quads each. V staging: key pairs, 8 cols each.
  const int skey = tid >> 3,        sseg = tid & 7;
  const int kd0  = sseg * 24;
  const int vkey = (tid >> 4) << 1; // 0,2,...,30
  const int vd0  = (tid & 15) * 8;  // dv block of 8

  const int kend = q0 + 128;        // causal bound for this block
  for (int kb = 0; kb < kend; kb += 32) {
    // ---- K tile: row-major [key][d], vector b64 stores ----
    const float* kvrow = kvup + (size_t)(tok0 + kb + skey) * 4096 + h * 256;
    const float* kprow = kpe + (size_t)(tok0 + kb + skey) * 64;
    float4 kq[6];
    #pragma unroll
    for (int t = 0; t < 6; ++t) {
      int d = kd0 + 4 * t;
      kq[t] = (d < 128) ? *(const float4*)(kvrow + d) : *(const float4*)(kprow + d - 128);
    }
    // ---- V tile transposed: pack (key, key+1) -> b32 stores ----
    const float* vr0 = kvup + (size_t)(tok0 + kb + vkey) * 4096 + h * 256 + 128 + vd0;
    const float* vr1 = vr0 + 4096;
    float4 va0 = *(const float4*)vr0;
    float4 va1 = *(const float4*)(vr0 + 4);
    float4 vb0 = *(const float4*)vr1;
    float4 vb1 = *(const float4*)(vr1 + 4);
    #pragma unroll
    for (int t = 0; t < 6; ++t)
      *(bf16x4*)&Ks[skey][kd0 + 4 * t] = cvt4(kq[t]);
    *(bf16x2*)&VsT[vd0 + 0][vkey] = pk2(va0.x, vb0.x);
    *(bf16x2*)&VsT[vd0 + 1][vkey] = pk2(va0.y, vb0.y);
    *(bf16x2*)&VsT[vd0 + 2][vkey] = pk2(va0.z, vb0.z);
    *(bf16x2*)&VsT[vd0 + 3][vkey] = pk2(va0.w, vb0.w);
    *(bf16x2*)&VsT[vd0 + 4][vkey] = pk2(va1.x, vb1.x);
    *(bf16x2*)&VsT[vd0 + 5][vkey] = pk2(va1.y, vb1.y);
    *(bf16x2*)&VsT[vd0 + 6][vkey] = pk2(va1.z, vb1.z);
    *(bf16x2*)&VsT[vd0 + 7][vkey] = pk2(va1.w, vb1.w);
    __syncthreads();

    if (kb <= qw + 15) {   // wave-uniform: this wave has unmasked keys here
      // ---- scores: two 16x16 tiles over 32 keys ----
      f32x8 sc[2] = { zero, zero };
      #pragma unroll
      for (int st = 0; st < 2; ++st) {
        #pragma unroll
        for (int c = 0; c < 6; ++c) {
          bf16x16 kf = ldsfrag(&Ks[st * 16 + ln][32 * c + 16 * hi]);
          sc[st] = WMMA_BF16(qf[c], kf, sc[st]);
        }
      }
      // ---- online softmax (fp32); rows m = v + 8*hi ----
      const int pcol = ln;              // contiguous position for key col ln
      #pragma unroll
      for (int v = 0; v < 8; ++v) {
        int qrow = qw + v + 8 * hi;
        float s0 = sc[0][v] * SCALE;
        float s1 = sc[1][v] * SCALE;
        if (kb + ln      > qrow) s0 = -1e30f;
        if (kb + 16 + ln > qrow) s1 = -1e30f;
        float mx = fmaxf(s0, s1);
        #pragma unroll
        for (int off = 8; off; off >>= 1) mx = fmaxf(mx, __shfl_xor(mx, off));
        float m_new = fmaxf(m_i[v], mx);
        float alpha = __expf(m_i[v] - m_new);
        float p0 = __expf(s0 - m_new);
        float p1 = __expf(s1 - m_new);
        float rs = p0 + p1;
        #pragma unroll
        for (int off = 8; off; off >>= 1) rs += __shfl_xor(rs, off);
        l_i[v] = l_i[v] * alpha + rs;
        m_i[v] = m_new;
        #pragma unroll
        for (int nt = 0; nt < 8; ++nt) acc[nt][v] *= alpha;
        Pb[wave][v + 8 * hi][pcol]     = (__bf16)p0;   // key ln
        Pb[wave][v + 8 * hi][pcol + 16] = (__bf16)p1;   // key 16+ln
      }
      // ---- P @ V ----
      bf16x16 pa = ldsfrag(&Pb[wave][ln][16 * hi]);
      #pragma unroll
      for (int nt = 0; nt < 8; ++nt) {
        bf16x16 vf = ldsfrag(&VsT[nt * 16 + ln][16 * hi]);
        acc[nt] = WMMA_BF16(pa, vf, acc[nt]);
      }
    }
    __syncthreads();
  }

  #pragma unroll
  for (int nt = 0; nt < 8; ++nt) {
    #pragma unroll
    for (int v = 0; v < 8; ++v) {
      int row = qw + v + 8 * hi;
      o[(size_t)(tok0 + row) * 2048 + h * 128 + nt * 16 + ln] = acc[nt][v] / l_i[v];
    }
  }
}

// =======================================================================
extern "C" void kernel_launch(void* const* d_in, const int* in_sizes, int n_in,
                              void* d_out, int out_size, void* d_ws, size_t ws_size,
                              hipStream_t stream) {
  constexpr int Bc = 2, Sc = 2048, Hc = 2048, NHc = 16;
  constexpr int QL = 512, KVL = 512, NOPEc = 128, ROPEc = 64, VD = 128, QKD = 192;
  constexpr int T = Bc * Sc;  // 4096 tokens

  const float* hidden = (const float*)d_in[0];
  const float* cosp   = (const float*)d_in[1];
  const float* sinp   = (const float*)d_in[2];
  const float* wq_a   = (const float*)d_in[3];
  const float* qnw    = (const float*)d_in[4];
  const float* wq_b   = (const float*)d_in[5];
  const float* wkv_a  = (const float*)d_in[6];
  const float* kvnw   = (const float*)d_in[7];
  const float* wkv_b  = (const float*)d_in[8];
  const float* wo     = (const float*)d_in[9];
  float* out = (float*)d_out;

  float* ws    = (float*)d_ws;
  float* q_a   = ws;                                     // T x 512
  float* kvb   = q_a   + (size_t)T * QL;                 // T x 576
  float* qfull = kvb   + (size_t)T * (KVL + ROPEc);      // T x 3072
  float* kvup  = qfull + (size_t)T * NHc * QKD;          // T x 4096
  float* kpe   = kvup  + (size_t)T * NHc * (NOPEc + VD); // T x 64
  float* attn  = kpe   + (size_t)T * ROPEc;              // T x 2048

  // 1) q_a = hidden @ wq_a   [4096,2048]x[2048,512]
  gemm_bf16_wmma<<<dim3(QL / 64, T / 64), 256, 0, stream>>>(
      hidden, wq_a, q_a, Hc, Hc, QL, QL);
  // 2) kv = hidden @ wkv_a   [4096,2048]x[2048,576]
  gemm_bf16_wmma<<<dim3((KVL + ROPEc) / 64, T / 64), 256, 0, stream>>>(
      hidden, wkv_a, kvb, Hc, Hc, KVL + ROPEc, KVL + ROPEc);
  // 3) RMSNorm (in place)
  rmsnorm512_kernel<<<T, 128, 0, stream>>>(q_a, qnw, q_a, QL, QL);
  rmsnorm512_kernel<<<T, 128, 0, stream>>>(kvb, kvnw, kvb, KVL + ROPEc, KVL + ROPEc);
  // 4) q = q_norm @ wq_b     [4096,512]x[512,3072]
  gemm_bf16_wmma<<<dim3(NHc * QKD / 64, T / 64), 256, 0, stream>>>(
      q_a, wq_b, qfull, QL, QL, NHc * QKD, NHc * QKD);
  // 5) kv_up = kv_norm @ wkv_b  [4096,512]x[512,4096] (lda=576 skips k_pe cols)
  gemm_bf16_wmma<<<dim3(NHc * (NOPEc + VD) / 64, T / 64), 256, 0, stream>>>(
      kvb, wkv_b, kvup, KVL, KVL + ROPEc, NHc * (NOPEc + VD), NHc * (NOPEc + VD));
  // 6) RoPE
  rope_q_kernel<<<(T * NHc * 32) / 256, 256, 0, stream>>>(qfull, cosp, sinp);
  rope_k_kernel<<<(T * 32) / 256, 256, 0, stream>>>(kvb, kpe, cosp, sinp);
  // 7) causal flash attention -> attn [4096, 2048]
  mla_flash_attn<<<dim3(Sc / 128, Bc * NHc), 256, 0, stream>>>(qfull, kvup, kpe, attn);
  // 8) out = attn @ wo       [4096,2048]x[2048,2048]
  gemm_bf16_wmma<<<dim3(Hc / 64, T / 64), 256, 0, stream>>>(
      attn, wo, out, NHc * VD, NHc * VD, Hc, Hc);
}
